// SelectAttachmentAtom_1967095022239
// MI455X (gfx1250) — matrix operand order repaired and checked by reading, not compiled, not run on hardware
//
#include <hip/hip_runtime.h>
#include <hip/hip_bf16.h>
#include <math.h>

// ---------------------------------------------------------------------------
// SelectAttachmentAtom MLP on MI455X (gfx1250): bf16 WMMA, f32 accumulate.
//   x[N,640] = [node_hidden(128) | c2[idx](256) | tm[idx](256)]
//   h1 = relu(x @ W1 + b1)   640 -> 256
//   h2 = relu(h1 @ W2 + b2)  256 -> 128
//   out = sigmoid(h2 @ W3 + b3)
// ---------------------------------------------------------------------------

typedef __attribute__((ext_vector_type(16))) __bf16 v16bf;
typedef __attribute__((ext_vector_type(8)))  __bf16 v8bf;
typedef __attribute__((ext_vector_type(4)))  __bf16 v4bf;
typedef __attribute__((ext_vector_type(8)))  float  v8f;

#define IN_DIM   640
#define H1_DIM   256
#define H2_DIM   128
#define M_TILE   64          // nodes per block
#define THREADS  128         // 4 waves (wave32)

// padded LDS strides (in bf16 elements) -> +4 dword shift per row, no bank conflicts
#define XS   648   // 640 + 8
#define H1S  264   // 256 + 8
#define H2S  136   // 128 + 8

#define NW1P (20 * 16 * 512)   // packed W1 bf16 elements (163840)
#define NW2P (8  * 8  * 512)   // packed W2 bf16 elements (32768)

// ---------------------------------------------------------------------------
// Pack W1/W2 (f32 row-major [K][N]) into bf16 WMMA B-fragment tiles.
// Tile = 32(K) x 16(N). Within a tile (512 elems), element p = lane*16 + q:
//   n = nt*16 + (lane & 15)
//   k = kt*32 + (lane >> 4)*8 + (q & 7) + (q >> 3)*16   (matches A table order)
// So each lane's 16 values are contiguous -> one v16bf load per lane.
// ---------------------------------------------------------------------------
__global__ __launch_bounds__(256) void pack_weights_kernel(
    const float* __restrict__ W1, const float* __restrict__ W2,
    __bf16* __restrict__ pW1, __bf16* __restrict__ pW2)
{
    int p = blockIdx.x * blockDim.x + threadIdx.x;
    if (p < NW1P) {
        int tile = p >> 9, r = p & 511;
        int lane = r >> 4, q = r & 15;
        int kt = tile >> 4, nt = tile & 15;            // 20 x 16 tiles
        int n = nt * 16 + (lane & 15);
        int k = kt * 32 + ((lane >> 4) << 3) + (q & 7) + ((q >> 3) << 4);
        pW1[p] = (__bf16)W1[k * H1_DIM + n];
    } else if (p < NW1P + NW2P) {
        int pp = p - NW1P;
        int tile = pp >> 9, r = pp & 511;
        int lane = r >> 4, q = r & 15;
        int kt = tile >> 3, nt = tile & 7;             // 8 x 8 tiles
        int n = nt * 16 + (lane & 15);
        int k = kt * 32 + ((lane >> 4) << 3) + (q & 7) + ((q >> 3) << 4);
        pW2[pp] = (__bf16)W2[k * H2_DIM + n];
    }
}

__device__ __forceinline__ int imin(int a, int b) { return a < b ? a : b; }

// ---------------------------------------------------------------------------
// Main fused MLP kernel.
// ---------------------------------------------------------------------------
__global__ __launch_bounds__(THREADS) void mlp_wmma_kernel(
    const float* __restrict__ nh,        // [N,128]
    const float* __restrict__ c2,        // [B,256]
    const float* __restrict__ tm,        // [B,256]
    const int*   __restrict__ bidx,      // [N]
    const float* __restrict__ b1,        // [256]
    const float* __restrict__ b2,        // [128]
    const float* __restrict__ W3,        // [128]
    const float* __restrict__ b3,        // [1]
    const __bf16* __restrict__ pW1,      // packed bf16 W1
    const __bf16* __restrict__ pW2,      // packed bf16 W2
    float* __restrict__ out, int N)
{
    extern __shared__ __align__(16) char smem_raw[];
    __bf16* xs  = (__bf16*)smem_raw;          // [64][XS]
    __bf16* h1s = xs  + M_TILE * XS;          // [64][H1S]
    __bf16* h2s = h1s + M_TILE * H1S;         // [64][H2S]

    const int tid  = threadIdx.x;
    const int base = blockIdx.x * M_TILE;

    // ---------------- load + convert x tile into LDS (bf16) ----------------
    // node_hiddens: 64 rows x 32 float4
    for (int e = tid; e < M_TILE * 32; e += THREADS) {
        int row = e >> 5, c4 = e & 31;
        int node = imin(base + row, N - 1);
        float4 v = ((const float4*)nh)[(size_t)node * 32 + c4];
        v4bf o = { (__bf16)v.x, (__bf16)v.y, (__bf16)v.z, (__bf16)v.w };
        *(v4bf*)&xs[row * XS + c4 * 4] = o;
    }
    // gathered c2 / tm: 64 rows x (64 + 64) float4
    for (int e = tid; e < M_TILE * 128; e += THREADS) {
        int row = e >> 7, p = e & 127;
        int node = imin(base + row, N - 1);
        int idx  = bidx[node];
        int q = p & 63;
        const float4* srcp = (p < 64) ? (const float4*)(c2 + (size_t)idx * 256)
                                      : (const float4*)(tm + (size_t)idx * 256);
        float4 v = srcp[q];
        int col = ((p < 64) ? 128 : 384) + q * 4;
        v4bf o = { (__bf16)v.x, (__bf16)v.y, (__bf16)v.z, (__bf16)v.w };
        *(v4bf*)&xs[row * XS + col] = o;
    }
    __syncthreads();

    const int wave = tid >> 5;
    const int lane = tid & 31;
    const int m    = lane & 15;     // M (A) / N (B,C,D) position within half-wave
    const int kh   = lane >> 4;     // half-wave selector

    const v16bf* pb1v = (const v16bf*)pW1;    // 32 v16bf per 32x16 tile
    const v16bf* pb2v = (const v16bf*)pW2;

    const v8f vzero = {0.f, 0.f, 0.f, 0.f, 0.f, 0.f, 0.f, 0.f};

    // ============================ layer 1: 640 -> 256 =======================
    // wave owns N-tiles [wave*4, wave*4+4), all 4 M-tiles.
    {
        v8f acc[4][4];
#pragma unroll
        for (int mt = 0; mt < 4; ++mt)
#pragma unroll
            for (int nt = 0; nt < 4; ++nt) acc[mt][nt] = vzero;

        for (int kt = 0; kt < IN_DIM / 32; ++kt) {
            v16bf a[4];
#pragma unroll
            for (int mt = 0; mt < 4; ++mt) {
                const __bf16* rp = &xs[(mt * 16 + m) * XS + kt * 32 + kh * 8];
                v8bf lo = *(const v8bf*)rp;          // K 0-7  (or 8-15)
                v8bf hi = *(const v8bf*)(rp + 16);   // K 16-23 (or 24-31)
                a[mt] = __builtin_shufflevector(lo, hi, 0,1,2,3,4,5,6,7,
                                                        8,9,10,11,12,13,14,15);
            }
            v16bf bf[4];
#pragma unroll
            for (int nt = 0; nt < 4; ++nt) {
                int tile = kt * 16 + wave * 4 + nt;
                bf[nt] = pb1v[tile * 32 + lane];
            }
#pragma unroll
            for (int mt = 0; mt < 4; ++mt)
#pragma unroll
                for (int nt = 0; nt < 4; ++nt)
                    acc[mt][nt] = __builtin_amdgcn_wmma_f32_16x16x32_bf16(
                        false, a[mt], false, bf[nt], (short)0, acc[mt][nt],
                        false, false);
        }
        // bias + relu + stage bf16 h1 in LDS
#pragma unroll
        for (int nt = 0; nt < 4; ++nt) {
            int n = (wave * 4 + nt) * 16 + m;
            float bias = b1[n];
#pragma unroll
            for (int mt = 0; mt < 4; ++mt)
#pragma unroll
                for (int i = 0; i < 8; ++i) {
                    float v = acc[mt][nt][i] + bias;
                    v = v > 0.f ? v : 0.f;
                    int row = mt * 16 + kh * 8 + i;   // C/D layout: M = i + 8*kh
                    h1s[row * H1S + n] = (__bf16)v;
                }
        }
    }
    __syncthreads();

    // ============================ layer 2: 256 -> 128 =======================
    // wave owns N-tiles [wave*2, wave*2+2)
    {
        v8f acc[4][2];
#pragma unroll
        for (int mt = 0; mt < 4; ++mt)
#pragma unroll
            for (int nt = 0; nt < 2; ++nt) acc[mt][nt] = vzero;

        for (int kt = 0; kt < H1_DIM / 32; ++kt) {
            v16bf a[4];
#pragma unroll
            for (int mt = 0; mt < 4; ++mt) {
                const __bf16* rp = &h1s[(mt * 16 + m) * H1S + kt * 32 + kh * 8];
                v8bf lo = *(const v8bf*)rp;
                v8bf hi = *(const v8bf*)(rp + 16);
                a[mt] = __builtin_shufflevector(lo, hi, 0,1,2,3,4,5,6,7,
                                                        8,9,10,11,12,13,14,15);
            }
            v16bf bf[2];
#pragma unroll
            for (int nt = 0; nt < 2; ++nt) {
                int tile = kt * 8 + wave * 2 + nt;
                bf[nt] = pb2v[tile * 32 + lane];
            }
#pragma unroll
            for (int mt = 0; mt < 4; ++mt)
#pragma unroll
                for (int nt = 0; nt < 2; ++nt)
                    acc[mt][nt] = __builtin_amdgcn_wmma_f32_16x16x32_bf16(
                        false, a[mt], false, bf[nt], (short)0, acc[mt][nt],
                        false, false);
        }
#pragma unroll
        for (int nt = 0; nt < 2; ++nt) {
            int n = (wave * 2 + nt) * 16 + m;
            float bias = b2[n];
#pragma unroll
            for (int mt = 0; mt < 4; ++mt)
#pragma unroll
                for (int i = 0; i < 8; ++i) {
                    float v = acc[mt][nt][i] + bias;
                    v = v > 0.f ? v : 0.f;
                    int row = mt * 16 + kh * 8 + i;
                    h2s[row * H2S + n] = (__bf16)v;
                }
        }
    }
    __syncthreads();

    // ===================== layer 3: 128 -> 1 + sigmoid ======================
    if (tid < M_TILE) {
        int node = base + tid;
        if (node < N) {
            float s = b3[0];
            const __bf16* hr = &h2s[tid * H2S];
#pragma unroll 8
            for (int k = 0; k < H2_DIM; ++k)
                s += (float)hr[k] * W3[k];
            out[node] = 1.f / (1.f + expf(-s));
        }
    }
}

// ---------------------------------------------------------------------------
extern "C" void kernel_launch(void* const* d_in, const int* in_sizes, int n_in,
                              void* d_out, int out_size, void* d_ws, size_t ws_size,
                              hipStream_t stream) {
    const float* nh   = (const float*)d_in[0];
    const float* c2   = (const float*)d_in[1];
    const float* tm   = (const float*)d_in[2];
    const int*   bidx = (const int*)  d_in[3];
    const float* W1   = (const float*)d_in[4];
    const float* b1   = (const float*)d_in[5];
    const float* W2   = (const float*)d_in[6];
    const float* b2   = (const float*)d_in[7];
    const float* W3   = (const float*)d_in[8];
    const float* b3   = (const float*)d_in[9];
    float* out        = (float*)d_out;

    const int N = in_sizes[0] / 128;

    __bf16* pW1 = (__bf16*)d_ws;
    __bf16* pW2 = pW1 + NW1P;

    // pack weights to bf16 WMMA fragment layout (L2-resident, tiny)
    {
        int total = NW1P + NW2P;
        int blocks = (total + 255) / 256;
        pack_weights_kernel<<<blocks, 256, 0, stream>>>(W1, W2, pW1, pW2);
    }

    // fused MLP
    {
        int blocks = (N + M_TILE - 1) / M_TILE;
        size_t smem = (size_t)(M_TILE * XS + M_TILE * H1S + M_TILE * H2S) * sizeof(__bf16);
        mlp_wmma_kernel<<<blocks, THREADS, smem, stream>>>(
            nh, c2, tm, bidx, b1, b2, W3, b3, pW1, pW2, out, N);
    }
}
